// ComplexityDecoderLayerV2_63660005262011
// MI455X (gfx1250) — matrix operand order repaired and checked by reading, not compiled, not run on hardware
//
#include <hip/hip_runtime.h>

typedef __attribute__((ext_vector_type(16))) __bf16 v16bf;
typedef __attribute__((ext_vector_type(8)))  __bf16 v8bf;
typedef __attribute__((ext_vector_type(8)))  float  v8f;
typedef __attribute__((ext_vector_type(4)))  unsigned int u32x4;
typedef __attribute__((ext_vector_type(8)))  int i32x8;
typedef __attribute__((ext_vector_type(4)))  int i32x4;

namespace {
constexpr int NTOK = 2048;     // N
constexpr int DM   = 2048;     // D
constexpr int NHEAD = 16;      // NH
constexpr int NKVH  = 4;       // NKV
constexpr int HDIM  = 128;     // HD
constexpr int NEXP  = 4;       // E
constexpr int EIDIM = 8192;    // EI = I/E
constexpr int CHUNK = NTOK / NEXP; // 512
constexpr float EPSV = 1e-6f;
}

// ---------------------------------------------------------------------------
// CDNA5 Tensor Data Mover: 2-D tile load global->LDS (ISA cdna5 ch.8 D#),
// probe-confirmed builtin (5-arg on clang-22/ROCm7.2, 6-arg on clang-23).
// ---------------------------------------------------------------------------
#if __has_builtin(__builtin_amdgcn_tensor_load_to_lds)
#define HAVE_TDM 1
// rows x width (elements, 2B each) tile at gaddr with row stride stride_elems,
// stored to LDS with optional padding: pad_interval/pad_amount are the ISA
// encodings (interval: 0=2DW,1=4,2=8,3=16,4=32,5=64; amount: n -> n+1 DWORDs).
static __device__ __forceinline__ void tdm_load_2d(const void* gaddr, void* lds,
                                                   unsigned rows, unsigned width,
                                                   unsigned stride_elems,
                                                   unsigned pad_interval,
                                                   unsigned pad_amount) {
    const unsigned lds_off =
        (unsigned)(unsigned long long)(__attribute__((address_space(3))) char*)lds;
    const unsigned long long ga = (unsigned long long)gaddr;
    u32x4 g0;
    g0[0] = 1u;                                          // count=1, no gather
    g0[1] = lds_off;                                     // lds_addr (bytes)
    g0[2] = (unsigned)(ga & 0xFFFFFFFFu);                // global_addr[31:0]
    g0[3] = (unsigned)((ga >> 32) & 0x01FFFFFFu)         // global_addr[56:32]
          | (2u << 30);                                  // type=2 ("image")
    i32x8 g1;
    unsigned w0 = (1u << 16);                            // data_size=1 -> 2 bytes
    if (pad_amount) w0 |= (1u << 20)                     // pad_enable
                        | (pad_interval << 22) | (pad_amount << 25);
    g1[0] = (int)w0;                                     // wg_mask=0 (no cluster)
    g1[1] = (int)((width & 0xFFFFu) << 16);              // tensor_dim0[15:0] @48
    g1[2] = (int)(((width >> 16) & 0xFFFFu) | ((rows & 0xFFFFu) << 16)); // d0 hi | d1 lo
    g1[3] = (int)(((rows >> 16) & 0xFFFFu) | ((width & 0xFFFFu) << 16)); // d1 hi | tile_dim0
    g1[4] = (int)(rows & 0xFFFFu);                       // tile_dim1; tile_dim2=0
    g1[5] = (int)stride_elems;                           // tensor_dim0_stride[31:0]
    g1[6] = 0;                                           // stride hi | dim1_stride lo
    g1[7] = 0;
    const i32x4 z4 = {0, 0, 0, 0};
#if defined(__clang_major__) && __clang_major__ >= 23
    const i32x8 z8 = {0, 0, 0, 0, 0, 0, 0, 0};
    __builtin_amdgcn_tensor_load_to_lds(g0, g1, z4, z4, z8, 0);
#else
    __builtin_amdgcn_tensor_load_to_lds(g0, g1, z4, z4, 0);
#endif
}
static __device__ __forceinline__ void tdm_wait() {
#if __has_builtin(__builtin_amdgcn_s_wait_tensorcnt)
    __builtin_amdgcn_s_wait_tensorcnt(0);
#else
    asm volatile("s_wait_tensorcnt 0x0" ::: "memory");
#endif
}
#else
#define HAVE_TDM 0
static __device__ __forceinline__ void tdm_wait() {}
#endif

// ---------------------------------------------------------------------------
// fragment helpers
// ---------------------------------------------------------------------------
static __device__ __forceinline__ v16bf ld_frag16(const __bf16* p_lo, const __bf16* p_hi) {
    v8bf lo = *(const v8bf*)p_lo;
    v8bf hi = *(const v8bf*)p_hi;
    return __builtin_shufflevector(lo, hi, 0,1,2,3,4,5,6,7,8,9,10,11,12,13,14,15);
}

static __device__ __forceinline__ v8f wmma_bf16(v16bf a, v16bf b, v8f c) {
    // D = A(16x32,bf16) * B(32x16,bf16) + C(16x16,f32)
    return __builtin_amdgcn_wmma_f32_16x16x32_bf16(false, a, false, b, (short)0, c, false, false);
}

// ---------------------------------------------------------------------------
// RMSNorm (f32 in) -> bf16 out, one block per row of width DM
// ---------------------------------------------------------------------------
__global__ __launch_bounds__(256)
void rmsnorm_bf16_kernel(const float* __restrict__ X, const float* __restrict__ W,
                         __bf16* __restrict__ Y) {
    const int r = blockIdx.x;
    const int tid = threadIdx.x;
    const float* x = X + (size_t)r * DM;
    float ss = 0.f;
    for (int c = tid; c < DM; c += 256) { float v = x[c]; ss += v * v; }
    #pragma unroll
    for (int o = 16; o; o >>= 1) ss += __shfl_xor(ss, o, 32);
    __shared__ float red[8];
    __shared__ float sscale;
    if ((tid & 31) == 0) red[tid >> 5] = ss;
    __syncthreads();
    if (tid == 0) {
        float t = 0.f;
        #pragma unroll
        for (int i = 0; i < 8; ++i) t += red[i];
        sscale = rsqrtf(t / (float)DM + EPSV);
    }
    __syncthreads();
    const float sc = sscale;
    for (int c = tid; c < DM; c += 256)
        Y[(size_t)r * DM + c] = (__bf16)(x[c] * sc * W[c]);
}

// ---------------------------------------------------------------------------
// Per-(token, head) RMSNorm + RoPE, f32 in -> bf16 out. 128 threads per block.
// ---------------------------------------------------------------------------
__global__ __launch_bounds__(128)
void qknorm_rope_kernel(const float* __restrict__ X, const float* __restrict__ W,
                        const int* __restrict__ pos, __bf16* __restrict__ Y, int Hn) {
    const int tok = blockIdx.x, h = blockIdx.y;
    const int i = threadIdx.x;                       // 0..127
    const float* x = X + ((size_t)tok * Hn + h) * HDIM;
    float v = x[i];
    float ss = v * v;
    #pragma unroll
    for (int o = 16; o; o >>= 1) ss += __shfl_xor(ss, o, 32);
    __shared__ float red[4];
    __shared__ float normed[HDIM];
    if ((i & 31) == 0) red[i >> 5] = ss;
    __syncthreads();
    const float tot = red[0] + red[1] + red[2] + red[3];
    const float sc = rsqrtf(tot / (float)HDIM + EPSV);
    normed[i] = v * sc * W[i];
    __syncthreads();
    if (i < 64) {
        const float p = (float)pos[tok];
        const float inv = __powf(10000.0f, -(float)i / 64.0f);
        const float ang = p * inv;
        const float cs = __cosf(ang), sn = __sinf(ang);
        const float x1 = normed[i], x2 = normed[i + 64];
        __bf16* y = Y + ((size_t)tok * Hn + h) * HDIM;
        y[i]      = (__bf16)(x1 * cs - x2 * sn);
        y[i + 64] = (__bf16)(x2 * cs + x1 * sn);
    }
}

// ---------------------------------------------------------------------------
// Generic bf16 WMMA GEMM: C[M,O] = A[M,K](bf16) * B[K,O](f32, converted on the fly)
// BM=128 BN=128 BK=32, 256 threads = 8 waves in a 2(M) x 4(N) grid; each wave
// owns a 64x32 strip = 4x2 accumulators -> 8 WMMAs per k-step per wave.
// Double-buffered LDS; A tile staged by the Tensor Data Mover when rows are
// not gathered (amap==null), else manual gather staging.
// ---------------------------------------------------------------------------
namespace {
constexpr int BM = 128, BN = 128, BK = 32, BKP = 48; // BKP: LDS row stride (96B)
}

__global__ __launch_bounds__(256)
void gemm_bf16_wmma_kernel(const __bf16* __restrict__ A, int lda, const int* __restrict__ amap,
                           const float* __restrict__ B, int ldb, int blayout, long long bestride,
                           float* __restrict__ Cf, __bf16* __restrict__ Cb, int ldc,
                           const int* __restrict__ cmap, const float* __restrict__ addend,
                           const float* __restrict__ colbias,
                           int chunk, int K) {
    __shared__ __bf16 As[2][BM * BKP];
    __shared__ __bf16 Bs[2][BN * BKP];

    const int tid = threadIdx.x;
    const int lane = tid & 31, wid = tid >> 5;
    const int wm = wid & 1, wn = wid >> 1;           // 2 x 4 wave grid
    const int e = blockIdx.z;
    const int mBase = e * chunk + blockIdx.x * BM;
    const int nBase = blockIdx.y * BN;
    const float* Be = B + (long long)e * bestride;

    v8f acc[4][2];
    const v8f vzero = {};
    #pragma unroll
    for (int s = 0; s < 4; ++s)
        #pragma unroll
        for (int t = 0; t < 2; ++t) acc[s][t] = vzero;

    // A stage (manual path): thread -> row tid/2, 16 contiguous bf16 at (tid&1)*16
    const int ar  = tid >> 1;
    const int acg = (tid & 1) * 16;
    int arow = mBase + ar;
    if (amap) arow = amap[arow];
    const __bf16* aptr = A + (long long)arow * lda + acg;

    auto stageA = [&](int buf, int k0) {
#if HAVE_TDM
        if (amap == nullptr) {
            // one TDM descriptor moves the whole 128x32 bf16 tile, padding each
            // 16-DW row with 8 DW to reproduce the BKP=48 LDS stride
            if (wid == 0)
                tdm_load_2d(A + (long long)mBase * lda + k0, &As[buf][0],
                            BM, BK, (unsigned)lda, /*interval 16DW*/3, /*pad 8DW*/7);
            return;
        }
#endif
        const __bf16* src = aptr + k0;
        __bf16* dst = &As[buf][ar * BKP + acg];
        *(uint4*)dst       = *(const uint4*)src;
        *(uint4*)(dst + 8) = *(const uint4*)(src + 8);
        __builtin_prefetch(src + 2 * BK, 0, 1);
    };

    auto stageB = [&](int buf, int k0) {
        if (blayout == 1) {
            // WT: B element [k][n] = W[n][k]; Bs[n][k], k contiguous (reg+convert)
            const int br = tid >> 1, bcg = (tid & 1) * 16;
            const float* bp = Be + (long long)(nBase + br) * ldb + k0 + bcg;
            float4 f0 = *(const float4*)bp;
            float4 f1 = *(const float4*)(bp + 4);
            float4 f2 = *(const float4*)(bp + 8);
            float4 f3 = *(const float4*)(bp + 12);
            __bf16 t0[8] = {(__bf16)f0.x, (__bf16)f0.y, (__bf16)f0.z, (__bf16)f0.w,
                            (__bf16)f1.x, (__bf16)f1.y, (__bf16)f1.z, (__bf16)f1.w};
            __bf16 t1[8] = {(__bf16)f2.x, (__bf16)f2.y, (__bf16)f2.z, (__bf16)f2.w,
                            (__bf16)f3.x, (__bf16)f3.y, (__bf16)f3.z, (__bf16)f3.w};
            *(uint4*)&Bs[buf][br * BKP + bcg]     = *(const uint4*)t0;
            *(uint4*)&Bs[buf][br * BKP + bcg + 8] = *(const uint4*)t1;
            __builtin_prefetch(bp + BK, 0, 1);
        } else {
            // WN: B element [k][n] = W[k][n]; transpose+convert into Bs[n][k]
            const int bk = tid & 31, bng = (tid >> 5) * 16;
            const float* bp = Be + (long long)(k0 + bk) * ldb + nBase + bng;
            float4 f0 = *(const float4*)bp;
            float4 f1 = *(const float4*)(bp + 4);
            float4 f2 = *(const float4*)(bp + 8);
            float4 f3 = *(const float4*)(bp + 12);
            float fv[16] = {f0.x, f0.y, f0.z, f0.w, f1.x, f1.y, f1.z, f1.w,
                            f2.x, f2.y, f2.z, f2.w, f3.x, f3.y, f3.z, f3.w};
            #pragma unroll
            for (int j = 0; j < 16; ++j) Bs[buf][(bng + j) * BKP + bk] = (__bf16)fv[j];
            __builtin_prefetch(Be + (long long)(k0 + BK + bk) * ldb + nBase + bng, 0, 1);
        }
    };

    // prologue
    stageA(0, 0);
    stageB(0, 0);

    int buf = 0;
    for (int k0 = 0; k0 < K; k0 += BK) {
#if HAVE_TDM
        if (amap == nullptr) tdm_wait();   // TDM writes must land before publish
#endif
        __syncthreads();                   // publish buf; buf^1 free to overwrite

        const int kn = k0 + BK;
        if (kn < K) { stageA(buf ^ 1, kn); stageB(buf ^ 1, kn); }

        // ---- fragments + 8 WMMAs ----
        const int akb = (lane >> 4) * 8;     // A: lane<16 K 0-7/16-23, lane>=16 K 8-15/24-31
        const int bkb = (lane >> 4) * 8;     // B: lane<16 K 0-7/16-23, lane>=16 K 8-15/24-31
        v16bf af[4];
        #pragma unroll
        for (int s = 0; s < 4; ++s) {
            const int r = wm * 64 + s * 16 + (lane & 15);
            af[s] = ld_frag16(&As[buf][r * BKP + akb], &As[buf][r * BKP + akb + 16]);
        }
        #pragma unroll
        for (int t = 0; t < 2; ++t) {
            const int n = wn * 32 + t * 16 + (lane & 15);
            v16bf bfrag = ld_frag16(&Bs[buf][n * BKP + bkb], &Bs[buf][n * BKP + bkb + 16]);
            #pragma unroll
            for (int s = 0; s < 4; ++s) acc[s][t] = wmma_bf16(af[s], bfrag, acc[s][t]);
        }
        buf ^= 1;
    }

    // ---- epilogue: C layout lane<16 -> rows j, lane>=16 -> rows j+8, col = lane%16 ----
    #pragma unroll
    for (int s = 0; s < 4; ++s) {
        const int mrow = wm * 64 + s * 16 + (lane >> 4) * 8;
        #pragma unroll
        for (int t = 0; t < 2; ++t) {
            const int nc = nBase + wn * 32 + t * 16 + (lane & 15);
            #pragma unroll
            for (int j = 0; j < 8; ++j) {
                const int m = mBase + mrow + j;
                const int orow = cmap ? cmap[m] : m;
                float v = acc[s][t][j];
                const long long off = (long long)orow * ldc + nc;
                if (addend)  v += addend[off];
                if (colbias) v += colbias[nc];
                if (Cb) Cb[off] = (__bf16)v; else Cf[off] = v;
            }
        }
    }
}

// ---------------------------------------------------------------------------
// WMMA flash attention (causal GQA). Block: 128 queries x 1 head, 8 waves,
// each wave owns 16 queries. K tile staged by TDM, V transposed manually.
// ---------------------------------------------------------------------------
namespace {
constexpr int KBLK = 64;
constexpr int KSP = 136;  // Ks row stride (elems), 272B
constexpr int VSP = 72;   // Vst row stride, 144B
constexpr int PSP = 48;   // Ps row stride, 96B
}

__global__ __launch_bounds__(256)
void flash_attn_kernel(const __bf16* __restrict__ Qb, const __bf16* __restrict__ Kb,
                       const __bf16* __restrict__ Vb, __bf16* __restrict__ Ob) {
    __shared__ __bf16 Ks[KBLK * KSP];        // [key][feat]
    __shared__ __bf16 Vst[HDIM * VSP];       // [feat][key] (transposed)
    __shared__ __bf16 Ps[8 * 16 * PSP];      // per-wave P staging (16x32)

    const int tid = threadIdx.x, lane = tid & 31, wid = tid >> 5;
    const int qbase = blockIdx.x * 128;
    const int head = blockIdx.y;
    const int kvh = head / (NHEAD / NKVH);
    const int qw = qbase + wid * 16;
    const float scale = 0.08838834764831845f;  // HD^-0.5

    // Q fragments: 4 chunks of K=32 over HDIM=128
    v16bf aq[4];
    {
        const int qtok = qw + (lane & 15);
        const __bf16* qp = Qb + (long long)qtok * (NHEAD * HDIM) + head * HDIM;
        const int kb8 = (lane >> 4) * 8;
        #pragma unroll
        for (int c = 0; c < 4; ++c)
            aq[c] = ld_frag16(qp + c * 32 + kb8, qp + c * 32 + 16 + kb8);
    }

    v8f oacc[8];
    const v8f vzero = {};
    #pragma unroll
    for (int f = 0; f < 8; ++f) oacc[f] = vzero;
    float mrun[8], lrun[8];
    #pragma unroll
    for (int j = 0; j < 8; ++j) { mrun[j] = -1e30f; lrun[j] = 0.f; }

    const int nblocks = (qbase + 128) / KBLK;  // causal bound
    for (int jb = 0; jb < nblocks; ++jb) {
        const int kb0 = jb * KBLK;
        // ---- stage K (row-major): TDM when available ----
#if HAVE_TDM
        if (wid == 0)
            tdm_load_2d(Kb + (long long)kb0 * (NKVH * HDIM) + kvh * HDIM, &Ks[0],
                        KBLK, HDIM, NKVH * HDIM, /*interval 64DW*/5, /*pad 4DW*/3);
#else
        {
            const int r = tid >> 2, cg = (tid & 3) * 32;
            const __bf16* kp = Kb + (long long)(kb0 + r) * (NKVH * HDIM) + kvh * HDIM + cg;
            uint4* dst = (uint4*)&Ks[r * KSP + cg];
            const uint4* kp4 = (const uint4*)kp;
            dst[0] = kp4[0]; dst[1] = kp4[1]; dst[2] = kp4[2]; dst[3] = kp4[3];
        }
#endif
        // ---- stage V transposed (reg staging) ----
        {
            const int vk = tid & 63, fg = (tid >> 6) * 32;
            const __bf16* vp = Vb + (long long)(kb0 + vk) * (NKVH * HDIM) + kvh * HDIM + fg;
            __bf16 vtmp[32];
            const uint4* vp4 = (const uint4*)vp;
            *(uint4*)&vtmp[0]  = vp4[0];
            *(uint4*)&vtmp[8]  = vp4[1];
            *(uint4*)&vtmp[16] = vp4[2];
            *(uint4*)&vtmp[24] = vp4[3];
            #pragma unroll
            for (int j = 0; j < 32; ++j) Vst[(fg + j) * VSP + vk] = vtmp[j];
        }
        tdm_wait();
        __syncthreads();

        // ---- two 32-key pairs per block ----
        #pragma unroll 1
        for (int pr = 0; pr < 2; ++pr) {
            const int kstart = kb0 + pr * 32;
            if (kstart > qw + 15) break;   // fully masked for this wave

            // S = Q K^T for two 16-key subtiles
            v8f s0 = vzero, s1 = vzero;
            #pragma unroll
            for (int c = 0; c < 4; ++c) {
                const int fb = c * 32 + (lane >> 4) * 8;
                const int kr0 = pr * 32 + (lane & 15);
                v16bf bk0 = ld_frag16(&Ks[kr0 * KSP + fb], &Ks[kr0 * KSP + fb + 16]);
                s0 = wmma_bf16(aq[c], bk0, s0);
                const int kr1 = pr * 32 + 16 + (lane & 15);
                v16bf bk1 = ld_frag16(&Ks[kr1 * KSP + fb], &Ks[kr1 * KSP + fb + 16]);
                s1 = wmma_bf16(aq[c], bk1, s1);
            }

            // Online softmax over 32 keys (row m lives on VGPR j across lanes of a half)
            float corr[8];
            __bf16* pw = &Ps[wid * 16 * PSP];
            #pragma unroll
            for (int j = 0; j < 8; ++j) {
                const int m = j + ((lane >> 4) << 3);
                const int qi = qw + m;
                const int kc = kstart + (lane & 15);
                float v0 = s0[j] * scale; if (kc > qi)      v0 = -1e30f;
                float v1 = s1[j] * scale; if (kc + 16 > qi) v1 = -1e30f;
                float mx = fmaxf(v0, v1);
                #pragma unroll
                for (int o = 8; o; o >>= 1) mx = fmaxf(mx, __shfl_xor(mx, o, 32));
                const float nm = fmaxf(mrun[j], mx);
                const float p0 = __expf(v0 - nm), p1 = __expf(v1 - nm);
                float rs = p0 + p1;
                #pragma unroll
                for (int o = 8; o; o >>= 1) rs += __shfl_xor(rs, o, 32);
                corr[j] = __expf(mrun[j] - nm);
                lrun[j] = lrun[j] * corr[j] + rs;
                mrun[j] = nm;
                pw[m * PSP + (lane & 15)]      = (__bf16)p0;
                pw[m * PSP + 16 + (lane & 15)] = (__bf16)p1;
            }
            // rescale O accumulators
            #pragma unroll
            for (int f = 0; f < 8; ++f)
                #pragma unroll
                for (int j = 0; j < 8; ++j) oacc[f][j] = oacc[f][j] * corr[j];

            // P (C-layout) -> A-fragment via wave-private LDS round trip
            const int prow = lane & 15;
            const int pk8 = (lane >> 4) * 8;
            v16bf pfrag = ld_frag16(&pw[prow * PSP + pk8], &pw[prow * PSP + pk8 + 16]);

            // O += P * V  (8 feature subtiles of 16)
            const int vkb = pr * 32 + (lane >> 4) * 8;
            #pragma unroll
            for (int f = 0; f < 8; ++f) {
                const int feat = f * 16 + (lane & 15);
                v16bf vfrag = ld_frag16(&Vst[feat * VSP + vkb], &Vst[feat * VSP + vkb + 16]);
                oacc[f] = wmma_bf16(pfrag, vfrag, oacc[f]);
            }
        }
        __syncthreads();
    }

    // ---- normalize + write bf16 (token order) ----
    #pragma unroll
    for (int f = 0; f < 8; ++f) {
        const int feat = f * 16 + (lane & 15);
        #pragma unroll
        for (int j = 0; j < 8; ++j) {
            const int m = j + ((lane >> 4) << 3);
            const int tok = qw + m;
            const float v = oacc[f][j] / lrun[j];
            Ob[(long long)tok * (NHEAD * HDIM) + head * HDIM + feat] = (__bf16)v;
        }
    }
}

// ---------------------------------------------------------------------------
// small elementwise kernels
// ---------------------------------------------------------------------------
__global__ __launch_bounds__(256)
void f32_to_bf16_kernel(const float* __restrict__ X, __bf16* __restrict__ Y, long long n) {
    long long i = (long long)blockIdx.x * 256 + threadIdx.x;
    if (i < n) Y[i] = (__bf16)X[i];
}

__global__ __launch_bounds__(256)
void mu_clip_kernel(const float* __restrict__ mu, float* __restrict__ muc) {
    int i = blockIdx.x * 256 + threadIdx.x;
    if (i < DM) muc[i] = fminf(fmaxf(mu[i], 0.f), 2.f);
}

__global__ __launch_bounds__(256)
void silu_mul_kernel(const float* __restrict__ gu, __bf16* __restrict__ act) {
    long long i = (long long)blockIdx.x * 256 + threadIdx.x;  // over NTOK*EIDIM
    const long long r = i >> 13;           // / 8192
    const long long c = i & (EIDIM - 1);
    const float g = gu[r * (2 * EIDIM) + c];
    const float u = gu[r * (2 * EIDIM) + EIDIM + c];
    const float s = g / (1.f + __expf(-g));
    act[i] = (__bf16)(s * u);
}

// ---------------------------------------------------------------------------
// host
// ---------------------------------------------------------------------------
extern "C" void kernel_launch(void* const* d_in, const int* in_sizes, int n_in,
                              void* d_out, int out_size, void* d_ws, size_t ws_size,
                              hipStream_t stream) {
    const float* hidden   = (const float*)d_in[0];
    const int*   positions= (const int*)d_in[1];
    const int*   sort_idx = (const int*)d_in[2];
    const float* in_ln_w  = (const float*)d_in[3];
    const float* q_proj_w = (const float*)d_in[4];
    const float* o_proj_w = (const float*)d_in[5];
    const float* k_w      = (const float*)d_in[6];
    const float* v_w      = (const float*)d_in[7];
    const float* q_norm_w = (const float*)d_in[8];
    const float* k_norm_w = (const float*)d_in[9];
    const float* mu       = (const float*)d_in[10];
    const float* mu_proj_w= (const float*)d_in[11];
    const float* post_ln_w= (const float*)d_in[12];
    const float* gate_up  = (const float*)d_in[13];
    const float* down_w   = (const float*)d_in[14];
    float* out_h  = (float*)d_out;                       // [N, D]
    float* out_mu = (float*)d_out + (size_t)NTOK * DM;   // [N, D]

    // workspace carve-up (~252 MB)
    char* p = (char*)d_ws;
    auto carve = [&](size_t bytes) { char* r = p; p += (bytes + 255) & ~(size_t)255; return r; };
    __bf16* h_bf16    = (__bf16*)carve((size_t)NTOK * DM * 2);
    float*  q_f32     = (float*) carve((size_t)NTOK * NHEAD * HDIM * 4);
    float*  k_f32     = (float*) carve((size_t)NTOK * NKVH * HDIM * 4);
    __bf16* v_bf16    = (__bf16*)carve((size_t)NTOK * NKVH * HDIM * 2);
    __bf16* q_bf16    = (__bf16*)carve((size_t)NTOK * NHEAD * HDIM * 2);
    __bf16* k_bf16    = (__bf16*)carve((size_t)NTOK * NKVH * HDIM * 2);
    __bf16* attn_bf16 = (__bf16*)carve((size_t)NTOK * NHEAD * HDIM * 2);
    float*  h2_f32    = (float*) carve((size_t)NTOK * DM * 4);
    __bf16* h2_bf16   = (__bf16*)carve((size_t)NTOK * DM * 2);
    __bf16* post_bf16 = (__bf16*)carve((size_t)NTOK * DM * 2);
    float*  muc       = (float*) carve((size_t)DM * 4);
    float*  gu_f32    = (float*) carve((size_t)NTOK * 2 * EIDIM * 4);
    __bf16* act_bf16  = (__bf16*)carve((size_t)NTOK * EIDIM * 2);
    (void)in_sizes; (void)n_in; (void)out_size; (void)ws_size;

    auto gemm = [&](const __bf16* A, int lda, const int* amap,
                    const float* B, int ldb, int blayout, long long bestride,
                    float* Cf, __bf16* Cb, int ldc, const int* cmap,
                    const float* addend, const float* colbias,
                    int chunk, int K, int O, int nE) {
        dim3 g(chunk / BM, O / BN, nE), b(256);
        gemm_bf16_wmma_kernel<<<g, b, 0, stream>>>(A, lda, amap, B, ldb, blayout, bestride,
                                                   Cf, Cb, ldc, cmap, addend, colbias, chunk, K);
    };

    // 1) h = rms_norm(hidden, input_ln_w) -> bf16
    rmsnorm_bf16_kernel<<<dim3(NTOK), dim3(256), 0, stream>>>(hidden, in_ln_w, h_bf16);

    // 2) q = routed_proj(h, q_proj_w)  (gather by sort_idx, scatter back)
    gemm(h_bf16, DM, sort_idx, q_proj_w, NHEAD * HDIM, /*WN*/0, (long long)DM * NHEAD * HDIM,
         q_f32, nullptr, NHEAD * HDIM, sort_idx, nullptr, nullptr, CHUNK, DM, NHEAD * HDIM, NEXP);

    // 3) k = h @ k_w.T ; 4) v = h @ v_w.T (v straight to bf16)
    gemm(h_bf16, DM, nullptr, k_w, DM, /*WT*/1, 0,
         k_f32, nullptr, NKVH * HDIM, nullptr, nullptr, nullptr, NTOK, DM, NKVH * HDIM, 1);
    gemm(h_bf16, DM, nullptr, v_w, DM, /*WT*/1, 0,
         nullptr, v_bf16, NKVH * HDIM, nullptr, nullptr, nullptr, NTOK, DM, NKVH * HDIM, 1);

    // 5) q/k RMSNorm + RoPE -> bf16
    qknorm_rope_kernel<<<dim3(NTOK, NHEAD), dim3(128), 0, stream>>>(q_f32, q_norm_w, positions, q_bf16, NHEAD);
    qknorm_rope_kernel<<<dim3(NTOK, NKVH),  dim3(128), 0, stream>>>(k_f32, k_norm_w, positions, k_bf16, NKVH);

    // 6) causal GQA flash attention
    flash_attn_kernel<<<dim3(NTOK / 128, NHEAD), dim3(256), 0, stream>>>(q_bf16, k_bf16, v_bf16, attn_bf16);

    // 7) h2 = residual + routed_proj(attn, o_proj_w)
    gemm(attn_bf16, NHEAD * HDIM, sort_idx, o_proj_w, DM, /*WN*/0, (long long)NHEAD * HDIM * DM,
         h2_f32, nullptr, DM, sort_idx, hidden, nullptr, CHUNK, NHEAD * HDIM, DM, NEXP);

    // 8) mu_current = clip(mu,0,2) + h2 @ mu_proj_w.T
    f32_to_bf16_kernel<<<dim3((NTOK * DM) / 256), dim3(256), 0, stream>>>(h2_f32, h2_bf16, (long long)NTOK * DM);
    mu_clip_kernel<<<dim3(DM / 256), dim3(256), 0, stream>>>(mu, muc);
    gemm(h2_bf16, DM, nullptr, mu_proj_w, DM, /*WT*/1, 0,
         out_mu, nullptr, DM, nullptr, nullptr, muc, NTOK, DM, DM, 1);

    // 9) post-LN, MoE MLP (sorted order), down-proj scatters + residual into d_out
    rmsnorm_bf16_kernel<<<dim3(NTOK), dim3(256), 0, stream>>>(h2_f32, post_ln_w, post_bf16);
    gemm(post_bf16, DM, sort_idx, gate_up, 2 * EIDIM, /*WN*/0, (long long)DM * 2 * EIDIM,
         gu_f32, nullptr, 2 * EIDIM, nullptr, nullptr, nullptr, CHUNK, DM, 2 * EIDIM, NEXP);
    silu_mul_kernel<<<dim3(((long long)NTOK * EIDIM) / 256), dim3(256), 0, stream>>>(gu_f32, act_bf16);
    gemm(act_bf16, EIDIM, nullptr, down_w, DM, /*WN*/0, (long long)EIDIM * DM,
         out_h, nullptr, DM, sort_idx, h2_f32, nullptr, CHUNK, EIDIM, DM, NEXP);
}
